// VectorQuantizer_5308579578058
// MI455X (gfx1250) — matrix operand order repaired and checked by reading, not compiled, not run on hardware
//
#include <hip/hip_runtime.h>
#include <hip/hip_bf16.h>

typedef __bf16 bf16_t;
typedef bf16_t v8bf  __attribute__((ext_vector_type(8)));
typedef bf16_t v16bf __attribute__((ext_vector_type(16)));
typedef float  v8f   __attribute__((ext_vector_type(8)));
typedef unsigned int u32x4 __attribute__((ext_vector_type(4)));
typedef int          i32x8 __attribute__((ext_vector_type(8)));
typedef int          i32x4 __attribute__((ext_vector_type(4)));

#define NVEC 32768   // 32*32*32 flattened spatial vectors
#define DIM  256
#define KB   8192
#define TK   32      // codebook entries per LDS tile (double-buffered)
#define LDSS 264     // padded bf16 row stride (528B) -> conflict-free b128 reads

// ---------------------------------------------------------------------------
// Kernel 1: codebook fp32 -> bf16, plus per-row squared norm.
// ---------------------------------------------------------------------------
__global__ __launch_bounds__(256)
void vq_prep_codebook(const float* __restrict__ cb, bf16_t* __restrict__ cbb,
                      float* __restrict__ enorm) {
    int wave = threadIdx.x >> 5;
    int lane = threadIdx.x & 31;
    int k = blockIdx.x * 8 + wave;
    const float* src = cb + (size_t)k * DIM + lane * 8;
    float4 a = *(const float4*)(src);
    float4 b = *(const float4*)(src + 4);
    v8bf o;
    o[0] = (bf16_t)a.x; o[1] = (bf16_t)a.y; o[2] = (bf16_t)a.z; o[3] = (bf16_t)a.w;
    o[4] = (bf16_t)b.x; o[5] = (bf16_t)b.y; o[6] = (bf16_t)b.z; o[7] = (bf16_t)b.w;
    *(v8bf*)(cbb + (size_t)k * DIM + lane * 8) = o;
    float ss = a.x*a.x + a.y*a.y + a.z*a.z + a.w*a.w
             + b.x*b.x + b.y*b.y + b.z*b.z + b.w*b.w;
    #pragma unroll
    for (int m = 16; m >= 1; m >>= 1) ss += __shfl_xor(ss, m, 32);
    if (lane == 0) enorm[k] = ss;
}

// ---------------------------------------------------------------------------
// Kernel 2: x NCHW fp32 -> x_flat [N, D] bf16 (transpose C <-> HW via LDS).
// ---------------------------------------------------------------------------
__global__ __launch_bounds__(256)
void vq_prep_x(const float* __restrict__ x, bf16_t* __restrict__ xb) {
    __shared__ float tile[32][33];
    int bid = blockIdx.x;
    int ct = bid & 7;          // channel tile (32 channels)
    int h  = (bid >> 3) & 31;
    int b  = bid >> 8;
    int w  = threadIdx.x & 31;
    int g  = threadIdx.x >> 5; // 0..7
    const float* src = x + (((size_t)b * 256 + ct * 32) * 32 + h) * 32 + w;
    #pragma unroll
    for (int i = 0; i < 4; i++) {
        int cc = g + i * 8;
        tile[cc][w] = src[(size_t)cc * 1024];
    }
    __syncthreads();
    int cl = threadIdx.x & 31;
    int w8 = threadIdx.x >> 5;
    bf16_t* dst = xb + ((size_t)b * 1024 + h * 32) * 256 + ct * 32 + cl;
    #pragma unroll
    for (int i = 0; i < 4; i++) {
        int ww = w8 + i * 8;
        dst[(size_t)ww * 256] = (bf16_t)tile[cl][ww];
    }
}

// ---------------------------------------------------------------------------
// TDM: DMA a TK x 256 bf16 codebook tile into LDS with a 528B padded row
// stride (pad 4 dwords after every 128 dwords), as one tensor_load_to_lds.
// D# group0: count=1 | lds_addr | global_addr | type=2.
// D# group1: data_size=4B, pad_enable, pad_interval=128dw, pad_amount=4dw,
//            tensor_dim0=128, tensor_dim1=8192, tile_dim0=128, tile_dim1=TK,
//            tensor_dim0_stride=128, tensor_dim1_stride=128.
// ---------------------------------------------------------------------------
__device__ __forceinline__ void tdm_load_tile(const bf16_t* gsrc, unsigned lds_off) {
    unsigned long long ga = (unsigned long long)(size_t)gsrc;
    u32x4 g0;
    g0[0] = 1u;                                   // count=1, user mode
    g0[1] = lds_off;                              // lds_addr (bytes)
    g0[2] = (unsigned)(ga & 0xffffffffu);         // global_addr[31:0]
    g0[3] = (unsigned)((ga >> 32) & 0x01ffffffu)  // global_addr[56:32]
            | (2u << 30);                         // type=2 (image)
    i32x8 g1;
    g1[0] = (2 << 16)        // data_size = 4B
          | (1 << 20)        // pad_enable
          | (6 << 22)        // pad_interval: 128 dwords
          | (3 << 25);       // pad_amount: 4 dwords (16B)
    g1[1] = (int)(128u << 16);   // tensor_dim0[15:0] = 128 dwords/row
    g1[2] = (int)(8192u << 16);  // tensor_dim1[15:0] = 8192 rows
    g1[3] = (int)(128u << 16);   // tile_dim0 = 128 dwords
    g1[4] = TK;                  // tile_dim1 = TK rows
    g1[5] = 128;                 // tensor_dim0_stride[31:0] = 128 dwords
    g1[6] = (int)(128u << 16);   // tensor_dim1_stride[15:0]
    g1[7] = 0;
    i32x4 gz4 = {0, 0, 0, 0};
    i32x8 gz8 = {0, 0, 0, 0, 0, 0, 0, 0};
    __builtin_amdgcn_tensor_load_to_lds(g0, g1, gz4, gz4, gz8, 0);
}

// ---------------------------------------------------------------------------
// Kernel 3: fused GEMM + argmax(x.e - 0.5|e|^2)  ==  argmin distance.
// 128 blocks x 8 waves; each wave owns TWO 16-row M tiles in registers.
// Codebook tiles stream through double-buffered LDS via the Tensor Data
// Mover (wave 0 issues tensor_load_to_lds, s_wait_tensorcnt + barrier),
// overlapping DMA of tile i+1 with WMMA compute on tile i. B fragments are
// software-pipelined (load j+1 before consuming j).
// ---------------------------------------------------------------------------
__global__ __launch_bounds__(256)
void vq_argmin(const bf16_t* __restrict__ xb, const bf16_t* __restrict__ cbb,
               const float* __restrict__ enorm, int* __restrict__ idx_out) {
    __shared__ bf16_t cbt[2][TK * LDSS];   // 2 x 16896 B

    int tid  = threadIdx.x;
    int lane = tid & 31;
    int wave = tid >> 5;
    int half = lane >> 4;   // K-half selector for A/B fragments
    int l16  = lane & 15;   // M index (A) / N column (B, C, D)
    int rowbase = blockIdx.x * 256 + wave * 32;

    // --- Load 2 x 8 A-fragments (32 rows x 256 dims) into registers.
    v16bf afr[2][8];
    #pragma unroll
    for (int t = 0; t < 2; t++) {
        const bf16_t* xrow = xb + (size_t)(rowbase + t * 16 + l16) * DIM;
        #pragma unroll
        for (int j = 0; j < 8; j++) {
            v8bf lo = *(const v8bf*)(xrow + j * 32 + half * 16);
            v8bf hi = *(const v8bf*)(xrow + j * 32 + half * 16 + 8);
            afr[t][j] = __builtin_shufflevector(lo, hi,
                        0,1,2,3,4,5,6,7,8,9,10,11,12,13,14,15);
        }
    }

    float best[16];
    int   bidx[16];
    #pragma unroll
    for (int r = 0; r < 16; r++) { best[r] = -3.4e38f; bidx[r] = 0; }

    // Prologue: DMA tile 0 into buffer 0.
    if (wave == 0) tdm_load_tile(cbb, (unsigned)(size_t)(&cbt[0][0]));

    for (int kt = 0; kt < KB; kt += TK) {
        int cur = (kt / TK) & 1;
        if (wave == 0) __builtin_amdgcn_s_wait_tensorcnt(0);
        __syncthreads();   // tile `cur` resident; buffer cur^1 fully consumed
        if (wave == 0 && (kt + TK) < KB)
            tdm_load_tile(cbb + (size_t)(kt + TK) * DIM,
                          (unsigned)(size_t)(&cbt[cur ^ 1][0]));

        // Hoisted score biases for this tile (L2-resident, latency hidden).
        float cbias0 = -0.5f * enorm[kt + l16];
        float cbias1 = -0.5f * enorm[kt + 16 + l16];

        #pragma unroll
        for (int sub = 0; sub < TK / 16; sub++) {
            float c0 = sub ? cbias1 : cbias0;
            v8f acc0 = {c0, c0, c0, c0, c0, c0, c0, c0};
            v8f acc1 = acc0;
            const bf16_t* brow = &cbt[cur][(sub * 16 + l16) * LDSS + half * 16];

            // Software-pipelined B fragments: fetch j+1 while j computes.
            v8bf blo = *(const v8bf*)(brow);
            v8bf bhi = *(const v8bf*)(brow + 8);
            v16bf bcur = __builtin_shufflevector(blo, bhi,
                         0,1,2,3,4,5,6,7,8,9,10,11,12,13,14,15);
            #pragma unroll
            for (int j = 0; j < 7; j++) {
                v8bf nlo = *(const v8bf*)(brow + (j + 1) * 32);
                v8bf nhi = *(const v8bf*)(brow + (j + 1) * 32 + 8);
                v16bf bnxt = __builtin_shufflevector(nlo, nhi,
                             0,1,2,3,4,5,6,7,8,9,10,11,12,13,14,15);
                acc0 = __builtin_amdgcn_wmma_f32_16x16x32_bf16(
                         false, afr[0][j], false, bcur, (short)0, acc0, false, false);
                acc1 = __builtin_amdgcn_wmma_f32_16x16x32_bf16(
                         false, afr[1][j], false, bcur, (short)0, acc1, false, false);
                bcur = bnxt;
            }
            acc0 = __builtin_amdgcn_wmma_f32_16x16x32_bf16(
                     false, afr[0][7], false, bcur, (short)0, acc0, false, false);
            acc1 = __builtin_amdgcn_wmma_f32_16x16x32_bf16(
                     false, afr[1][7], false, bcur, (short)0, acc1, false, false);

            int kg = kt + sub * 16 + l16;
            #pragma unroll
            for (int r = 0; r < 8; r++) {
                if (acc0[r] > best[r])     { best[r]     = acc0[r]; bidx[r]     = kg; }
                if (acc1[r] > best[8 + r]) { best[8 + r] = acc1[r]; bidx[8 + r] = kg; }
            }
        }
    }

    // Reduce across the 16 lanes sharing each row set (xor masks stay in-half).
    #pragma unroll
    for (int r = 0; r < 16; r++) {
        #pragma unroll
        for (int msk = 8; msk >= 1; msk >>= 1) {
            float ob = __shfl_xor(best[r], msk, 32);
            int   oi = __shfl_xor(bidx[r], msk, 32);
            if (ob > best[r] || (ob == best[r] && oi < bidx[r])) {
                best[r] = ob; bidx[r] = oi;
            }
        }
    }
    if (l16 == 0) {
        #pragma unroll
        for (int t = 0; t < 2; t++) {
            int rb = rowbase + t * 16 + half * 8;  // lanes 0/16: rows 0-7 / 8-15
            #pragma unroll
            for (int r = 0; r < 8; r++) idx_out[rb + r] = bidx[t * 8 + r];
        }
    }
}

// ---------------------------------------------------------------------------
// Kernel 4: gather fp32 codebook rows -> x_st (NCHW), float indices output,
// and per-block partial sums of (xq - x)^2.
// ---------------------------------------------------------------------------
__global__ __launch_bounds__(256)
void vq_gather(const float* __restrict__ x, const float* __restrict__ cb,
               const int* __restrict__ idx, float* __restrict__ out_xst,
               float* __restrict__ out_idx, float* __restrict__ partial) {
    int o = blockIdx.x * 256 + threadIdx.x;   // < 8388608
    int b  = o >> 18;
    int c  = (o >> 10) & 255;
    int hw = o & 1023;
    int n  = (b << 10) + hw;
    int k  = idx[n];
    float xq = cb[(size_t)k * DIM + c];
    float xo = x[o];
    out_xst[o] = xq;                       // x + sg(xq - x) == xq numerically
    if (c == 0) out_idx[n] = (float)k;
    float d = xq - xo;
    float ds = d * d;
    #pragma unroll
    for (int msk = 16; msk >= 1; msk >>= 1) ds += __shfl_xor(ds, msk, 32);
    __shared__ float wsum[8];
    if ((threadIdx.x & 31) == 0) wsum[threadIdx.x >> 5] = ds;
    __syncthreads();
    if (threadIdx.x == 0) {
        float s = 0.f;
        #pragma unroll
        for (int i = 0; i < 8; i++) s += wsum[i];
        partial[blockIdx.x] = s;
    }
}

// ---------------------------------------------------------------------------
// Kernel 5: deterministic final loss reduction.
// loss = (1 + BETA) * mean((xq - x)^2), BETA = 0.25.
// ---------------------------------------------------------------------------
__global__ __launch_bounds__(256)
void vq_loss_reduce(const float* __restrict__ partial, float* __restrict__ out_loss) {
    float s = 0.f;
    for (int i = threadIdx.x; i < 32768; i += 256) s += partial[i];
    #pragma unroll
    for (int msk = 16; msk >= 1; msk >>= 1) s += __shfl_xor(s, msk, 32);
    __shared__ float wsum[8];
    if ((threadIdx.x & 31) == 0) wsum[threadIdx.x >> 5] = s;
    __syncthreads();
    if (threadIdx.x == 0) {
        float t = 0.f;
        #pragma unroll
        for (int i = 0; i < 8; i++) t += wsum[i];
        *out_loss = t * (1.25f / 8388608.0f);
    }
}

// ---------------------------------------------------------------------------
extern "C" void kernel_launch(void* const* d_in, const int* in_sizes, int n_in,
                              void* d_out, int out_size, void* d_ws, size_t ws_size,
                              hipStream_t stream) {
    const float* x  = (const float*)d_in[0];   // [32,256,32,32]
    const float* cb = (const float*)d_in[1];   // [8192,256]

    char* ws = (char*)d_ws;
    bf16_t* xb    = (bf16_t*)(ws);                    // 16 MB
    bf16_t* cbb   = (bf16_t*)(ws + 16777216);         //  4 MB
    float*  enorm = (float*) (ws + 20971520);         // 32 KB
    int*    idxw  = (int*)   (ws + 21004288);         // 128 KB
    float*  part  = (float*) (ws + 21135360);         // 128 KB

    float* out      = (float*)d_out;
    float* out_xst  = out;                 // 8388608 floats (NCHW)
    float* out_loss = out + 8388608;       // 1 float
    float* out_idx  = out + 8388609;       // 32768 floats

    vq_prep_codebook<<<KB / 8, 256, 0, stream>>>(cb, cbb, enorm);
    vq_prep_x       <<<8192,  256, 0, stream>>>(x, xb);
    vq_argmin       <<<NVEC / 256, 256, 0, stream>>>(xb, cbb, enorm, idxw);
    vq_gather       <<<32768, 256, 0, stream>>>(x, cb, idxw, out_xst, out_idx, part);
    vq_loss_reduce  <<<1,     256, 0, stream>>>(part, out_loss);
}
